// MultiHeadAttention_53240414601281
// MI455X (gfx1250) — matrix operand order repaired and checked by reading, not compiled, not run on hardware
//
#include <hip/hip_runtime.h>
#include <hip/hip_bf16.h>
#include <cstdint>

typedef __bf16 bf16;
typedef __attribute__((ext_vector_type(16))) __bf16 v16bf;
typedef __attribute__((ext_vector_type(8)))  float  v8f;

struct __align__(16) U4 { unsigned x, y, z, w; };

union BfFrag {
    v16bf v;
    unsigned u[8];
    U4 q[2];
};

// A-matrix (16x32 bf16) per-lane K base for VGPR v (ISA 7.12.2):
// lanes 0-15: M=lane, VGPR0..3 -> K pairs {0..7}; VGPR4..7 -> K pairs {16..23}
// lanes 16-31: same M, K offset +8 in both groups
__device__ __forceinline__ int a_kbase(int lane, int v) {
    const int hi = (lane >= 16) ? 8 : 0;
    return (v < 4) ? (hi + 2 * v) : (16 + hi + 2 * (v - 4));
}

// B (32x16 bf16): lane = N col (lane&15), element s -> K in A's table order; p = K-row 0, ld = row stride
__device__ __forceinline__ v16bf load_b_frag(const bf16* __restrict__ p, int ld) {
    const int l = threadIdx.x & 31;
    const int hk = (l >> 4) << 3, n = l & 15;
    v16bf r;
#pragma unroll
    for (int s = 0; s < 16; ++s) {
        const int k = (s < 8) ? (hk + s) : (16 + hk + (s - 8));
        r[s] = p[(size_t)k * ld + n];
    }
    return r;
}

__device__ __forceinline__ v8f wmma_bf16(v16bf a, v16bf b, v8f c) {
    return __builtin_amdgcn_wmma_f32_16x16x32_bf16(
        false, a, false, b, (short)0, c, false, false);
}

// CDNA5 async global->LDS copy (ASYNCcnt domain). lds_addr = wave-relative LDS
// byte address (low 32 bits of flat shared address), 16 bytes per active lane.
__device__ __forceinline__ void async_b128_to_lds(unsigned lds_addr, const bf16* gptr) {
    asm volatile("global_load_async_to_lds_b128 %0, %1, off"
                 :: "v"(lds_addr), "v"(gptr)
                 : "memory");
}

__device__ __forceinline__ float rmax16(float x) {
#pragma unroll
    for (int off = 1; off < 16; off <<= 1)
        x = fmaxf(x, __shfl_xor(x, off, 32));
    return x;
}

__device__ __forceinline__ float rsum16(float x) {
#pragma unroll
    for (int off = 1; off < 16; off <<= 1)
        x += __shfl_xor(x, off, 32);
    return x;
}

// ---------------------------------------------------------------------------
// fp32 -> bf16 conversion
// ---------------------------------------------------------------------------
__global__ void cvt_f32_bf16(const float* __restrict__ in, bf16* __restrict__ out, int n) {
    int i = blockIdx.x * blockDim.x + threadIdx.x;
    if (i < n) out[i] = (bf16)in[i];
}

// ---------------------------------------------------------------------------
// WMMA GEMM: C[M,N] = A[M,K] bf16 row-major * B[K,N] bf16 row-major + bias.
// 256 threads = 8 waves; wave w owns rows [bx*128 + w*16, +16), cols [by*64, +64).
// B tiles (32x64 = 4KB/K-step) staged via async global->LDS through a 4-slot
// ring with a depth-2 pipeline (2 tiles in flight, ONE barrier per K-step).
// A fragments software-pipelined in registers. All four B fragments are pinned
// live so the DS reads clause under a single s_wait_dscnt before 4 WMMAs.
// Epilogue modes: OutB bf16 row-major | OutT bf16 per-head transpose
// [B,H,dk,S] (S=2048, H=16) | OutF fp32 row-major.
// ---------------------------------------------------------------------------
__global__ __launch_bounds__(256) void gemm_wmma(
    const bf16* __restrict__ A, const bf16* __restrict__ B,
    const float* __restrict__ bias,
    bf16* __restrict__ OutB, bf16* __restrict__ OutT, float* __restrict__ OutF,
    int M, int N, int K)
{
    __shared__ __align__(16) bf16 ldsB[4][32 * 64];   // 4 x 4KB ring

    const int tid  = threadIdx.x;
    const int lane = tid & 31;
    const int wave = tid >> 5;
    const int m0 = blockIdx.x * 128 + wave * 16;
    const int n0 = blockIdx.y * 64;

    const bf16* arow = A + (size_t)(m0 + (lane & 15)) * K;

    // cooperative B staging: thread tid copies 16 bytes of the 4KB tile
    const int bk = tid >> 3;          // K row within tile (0..31)
    const int bn = (tid & 7) * 8;     // N col within tile (elements)
    const unsigned lds_base = (unsigned)(uintptr_t)(&ldsB[0][0]) + (unsigned)tid * 16u;

    const int nk = K >> 5;

    // prologue: DMA tiles 0 and 1, load A fragment 0
    async_b128_to_lds(lds_base + 0 * 4096u, B + (size_t)bk * N + n0 + bn);
    async_b128_to_lds(lds_base + 1 * 4096u, B + (size_t)(32 + bk) * N + n0 + bn);
    BfFrag fa;
#pragma unroll
    for (int v = 0; v < 8; ++v)
        fa.u[v] = *reinterpret_cast<const unsigned*>(arow + a_kbase(lane, v));

    v8f acc[4] = {};
    for (int i = 0; i < nk; ++i) {
        BfFrag fan = fa;
        // kick off tile i+2 DMA and A fragment i+1 loads before the waits so
        // the current K-step's 4 WMMAs overlap both copies
        if (i + 2 < nk)
            async_b128_to_lds(lds_base + (unsigned)((i + 2) & 3) * 4096u,
                              B + (size_t)(((i + 2) << 5) + bk) * N + n0 + bn);
        if (i + 1 < nk) {
            const int knext = (i + 1) << 5;
#pragma unroll
            for (int v = 0; v < 8; ++v)
                fan.u[v] = *reinterpret_cast<const unsigned*>(arow + knext + a_kbase(lane, v));
        }
        // in-order completion: <=2 outstanding means tile i has landed
        if (i + 2 < nk)      asm volatile("s_wait_asynccnt 2" ::: "memory");
        else if (i + 1 < nk) asm volatile("s_wait_asynccnt 1" ::: "memory");
        else                 asm volatile("s_wait_asynccnt 0" ::: "memory");
        __syncthreads();   // single barrier per K-step (4-slot ring is skew-safe)

        const bf16* brow = &ldsB[i & 3][0];
        v16bf b0 = load_b_frag(brow, 64);
        v16bf b1 = load_b_frag(brow + 16, 64);
        v16bf b2 = load_b_frag(brow + 32, 64);
        v16bf b3 = load_b_frag(brow + 48, 64);
        // pin all four fragments live: forces the 64 strided 16-bit DS loads to clause
        // under ONE s_wait_dscnt, then 4 back-to-back WMMAs
        asm volatile("" : "+v"(b0), "+v"(b1), "+v"(b2), "+v"(b3));
        acc[0] = wmma_bf16(fa.v, b0, acc[0]);
        acc[1] = wmma_bf16(fa.v, b1, acc[1]);
        acc[2] = wmma_bf16(fa.v, b2, acc[2]);
        acc[3] = wmma_bf16(fa.v, b3, acc[3]);
        fa = fan;
    }

    // C layout: VGPR r, lane L -> M = r + (L>=16 ? 8 : 0), N = L&15
    const int hi8 = (lane >> 4) << 3;
#pragma unroll
    for (int t = 0; t < 4; ++t) {
#pragma unroll
        for (int r = 0; r < 8; ++r) {
            const int mm = m0 + r + hi8;
            const int nn = n0 + t * 16 + (lane & 15);
            const float o = acc[t][r] + bias[nn];
            if (OutF) OutF[(size_t)mm * N + nn] = o;
            if (OutB) OutB[(size_t)mm * N + nn] = (bf16)o;
            if (OutT) {
                const int bb = mm >> 11, ss = mm & 2047;   // S = 2048
                const int hh = nn >> 6,  dd = nn & 63;     // dk = 64
                OutT[((size_t)((bb * 16 + hh) * 64 + dd)) * 2048 + ss] = (bf16)o;
            }
        }
    }
}

// ---------------------------------------------------------------------------
// Flash attention: one wave per (b, h, 16-query tile).
// Q:  bf16 [B,S,1024] head-interleaved          -> A fragments (resident)
// Kt: bf16 [B,H,64,2048] per-head transpose     -> B fragments, contiguous
// V:  bf16 [B,S,1024]                           -> B fragments, contiguous
// K fragments double-buffered across chunks; V fragments prefetched before the
// softmax VALU block so global latency hides under exp/shuffle work.
// ---------------------------------------------------------------------------
__global__ __launch_bounds__(32) void flash_attn(
    const bf16* __restrict__ Q, const bf16* __restrict__ Kt,
    const bf16* __restrict__ V, bf16* __restrict__ Ctx)
{
    __shared__ __align__(16) bf16 lds_p[16 * 32];

    const int lane = threadIdx.x;
    const int q0 = blockIdx.x * 16;
    const int h  = blockIdx.y;
    const int b  = blockIdx.z;
    const int m   = lane & 15;
    const int hi8 = (lane >> 4) << 3;

    // Q A-fragments for d=0..31 / d=32..63, resident for the whole row
    const bf16* qrow = Q + ((size_t)(b * 2048 + q0 + m) * 1024 + h * 64);
    v16bf aq0, aq1;
    {
        BfFrag f0, f1;
#pragma unroll
        for (int v = 0; v < 8; ++v) {
            const int k = a_kbase(lane, v);
            f0.u[v] = *reinterpret_cast<const unsigned*>(qrow + k);
            f1.u[v] = *reinterpret_cast<const unsigned*>(qrow + 32 + k);
        }
        aq0 = f0.v; aq1 = f1.v;
    }

    float mrow[8], lrow[8];
    v8f acc[4] = {};
#pragma unroll
    for (int r = 0; r < 8; ++r) { mrow[r] = -1e30f; lrow[r] = 0.0f; }

    const bf16* ktbase = Kt + (size_t)((b * 16 + h) * 64) * 2048;
    const bf16* vbase  = V + ((size_t)(b * 2048) * 1024 + h * 64);
    const float scale = 0.125f;                 // 1/sqrt(64)
    const int nchunk = ((q0 + 15) >> 5) + 1;    // causal: skip past the diagonal

    // K fragments for chunk 0
    v16bf bk[4];
    {
        const bf16* p0 = ktbase;
        const bf16* p1 = ktbase + (size_t)32 * 2048;
        bk[0] = load_b_frag(p0, 2048);      bk[1] = load_b_frag(p1, 2048);
        bk[2] = load_b_frag(p0 + 16, 2048); bk[3] = load_b_frag(p1 + 16, 2048);
    }

    for (int c = 0; c < nchunk; ++c) {
        const int j0 = c << 5;

        // ---- scores: S = Q(16x64) . K^T, two 16-key tiles ----
        v8f s0 = {}, s1 = {};
        s0 = wmma_bf16(aq0, bk[0], s0);
        s0 = wmma_bf16(aq1, bk[1], s0);
        s1 = wmma_bf16(aq0, bk[2], s1);
        s1 = wmma_bf16(aq1, bk[3], s1);

        // ---- prefetch next chunk's K and this chunk's V fragments ----
        if (c + 1 < nchunk) {
            const int jn = j0 + 32;
            const bf16* p0 = ktbase + jn;
            const bf16* p1 = ktbase + (size_t)32 * 2048 + jn;
            bk[0] = load_b_frag(p0, 2048);      bk[1] = load_b_frag(p1, 2048);
            bk[2] = load_b_frag(p0 + 16, 2048); bk[3] = load_b_frag(p1 + 16, 2048);
        }
        v16bf bv[4];
#pragma unroll
        for (int t = 0; t < 4; ++t)
            bv[t] = load_b_frag(vbase + (size_t)j0 * 1024 + t * 16, 1024);

        // ---- causal mask + scale ----
        float p[2][8];
#pragma unroll
        for (int r = 0; r < 8; ++r) {
            const int q_abs = q0 + r + hi8;
            const int ja0 = j0 + m, ja1 = j0 + 16 + m;
            p[0][r] = (ja0 <= q_abs) ? s0[r] * scale : -1e30f;
            p[1][r] = (ja1 <= q_abs) ? s1[r] * scale : -1e30f;
        }

        // ---- online softmax (row lives across one 16-lane half) ----
#pragma unroll
        for (int r = 0; r < 8; ++r) {
            const float cm = rmax16(fmaxf(p[0][r], p[1][r]));
            const float mnew  = fmaxf(mrow[r], cm);
            const float alpha = __expf(mrow[r] - mnew);
            const float e0 = __expf(p[0][r] - mnew);
            const float e1 = __expf(p[1][r] - mnew);
            lrow[r] = lrow[r] * alpha + rsum16(e0 + e1);
            mrow[r] = mnew;
#pragma unroll
            for (int t = 0; t < 4; ++t) acc[t][r] *= alpha;
            const int q = r + hi8;
            lds_p[q * 32 + m]      = (bf16)e0;
            lds_p[q * 32 + 16 + m] = (bf16)e1;
        }
        __syncthreads();

        // ---- reload P as A fragment (16x32) ----
        v16bf ap;
        {
            BfFrag f;
#pragma unroll
            for (int v = 0; v < 8; ++v)
                f.u[v] = *reinterpret_cast<const unsigned*>(&lds_p[m * 32 + a_kbase(lane, v)]);
            ap = f.v;
        }
        __syncthreads();

        // ---- ctx += P @ V (4 d-tiles) ----
#pragma unroll
        for (int t = 0; t < 4; ++t)
            acc[t] = wmma_bf16(ap, bv[t], acc[t]);
    }

    // ---- epilogue: normalize, store bf16 ctx head-interleaved ----
#pragma unroll
    for (int r = 0; r < 8; ++r) {
        const float inv = 1.0f / lrow[r];
        const int q_abs = q0 + r + hi8;
#pragma unroll
        for (int t = 0; t < 4; ++t) {
            const float o = acc[t][r] * inv;
            Ctx[(size_t)(b * 2048 + q_abs) * 1024 + h * 64 + t * 16 + m] = (bf16)o;
        }
    }
}

// ---------------------------------------------------------------------------
// Launcher
// ---------------------------------------------------------------------------
extern "C" void kernel_launch(void* const* d_in, const int* in_sizes, int n_in,
                              void* d_out, int out_size, void* d_ws, size_t ws_size,
                              hipStream_t stream) {
    (void)in_sizes; (void)n_in; (void)out_size; (void)ws_size;

    const float* x  = (const float*)d_in[0];
    // d_in[1] = causal mask (implicit in flash_attn)
    const float* Wq = (const float*)d_in[2];
    const float* bq = (const float*)d_in[3];
    const float* Wk = (const float*)d_in[4];
    const float* bk = (const float*)d_in[5];
    const float* Wv = (const float*)d_in[6];
    const float* bv = (const float*)d_in[7];
    const float* Wo = (const float*)d_in[8];
    const float* bo = (const float*)d_in[9];
    float* out = (float*)d_out;

    const int M = 4096;   // B*S
    const int D = 1024;

    char* ws = (char*)d_ws;
    const size_t XB = (size_t)M * D * 2;     // 8 MB
    const size_t WB = (size_t)D * D * 2;     // 2 MB
    bf16* xb   = (bf16*)(ws);
    bf16* wqb  = (bf16*)(ws + XB);
    bf16* wkb  = (bf16*)(ws + XB + WB);
    bf16* wvb  = (bf16*)(ws + XB + 2 * WB);
    bf16* wob  = (bf16*)(ws + XB + 3 * WB);
    bf16* qb   = (bf16*)(ws + XB + 4 * WB);
    bf16* kt   = (bf16*)(ws + 2 * XB + 4 * WB);
    bf16* vb   = (bf16*)(ws + 3 * XB + 4 * WB);
    bf16* ctxb = (bf16*)(ws + 4 * XB + 4 * WB);

    // 1) fp32 -> bf16
    cvt_f32_bf16<<<(M * D + 255) / 256, 256, 0, stream>>>(x, xb, M * D);
    cvt_f32_bf16<<<(D * D + 255) / 256, 256, 0, stream>>>(Wq, wqb, D * D);
    cvt_f32_bf16<<<(D * D + 255) / 256, 256, 0, stream>>>(Wk, wkb, D * D);
    cvt_f32_bf16<<<(D * D + 255) / 256, 256, 0, stream>>>(Wv, wvb, D * D);
    cvt_f32_bf16<<<(D * D + 255) / 256, 256, 0, stream>>>(Wo, wob, D * D);

    // 2) projections
    dim3 ggrid(M / 128, D / 64);
    gemm_wmma<<<ggrid, 256, 0, stream>>>(xb, wqb, bq, qb, nullptr, nullptr, M, D, D);
    gemm_wmma<<<ggrid, 256, 0, stream>>>(xb, wkb, bk, nullptr, kt, nullptr, M, D, D);
    gemm_wmma<<<ggrid, 256, 0, stream>>>(xb, wvb, bv, vb, nullptr, nullptr, M, D, D);

    // 3) causal flash attention
    flash_attn<<<dim3(2048 / 16, 16, 2), 32, 0, stream>>>(qb, kt, vb, ctxb);

    // 4) output projection (fp32 result)
    gemm_wmma<<<ggrid, 256, 0, stream>>>(ctxb, wob, bo, nullptr, nullptr, out, M, D, D);
}
